// Model_52381421142390
// MI455X (gfx1250) — compile-verified, matrix operand order checked by reading, not run on hardware
//
#include <hip/hip_runtime.h>
#include <hip/hip_bf16.h>
#include <stdint.h>

typedef __attribute__((ext_vector_type(16))) _Float16 v16h;
typedef __attribute__((ext_vector_type(8)))  _Float16 v8h;
typedef __attribute__((ext_vector_type(8)))  float    v8f;
typedef unsigned int u32x4 __attribute__((ext_vector_type(4)));
typedef int          i32x8 __attribute__((ext_vector_type(8)));
typedef int          i32x4 __attribute__((ext_vector_type(4)));

// address-space-qualified element types for the async-LDS builtins
typedef __attribute__((address_space(1))) i32x4 g_v4i;   // global int4
typedef __attribute__((address_space(3))) i32x4 l_v4i;   // LDS int4
typedef __attribute__((address_space(1))) int   g_i32;   // global int
typedef __attribute__((address_space(3))) int   l_i32;   // LDS int

#ifndef __has_builtin
#define __has_builtin(x) 0
#endif

#if __has_builtin(__builtin_amdgcn_global_load_async_to_lds_b128) && \
    __has_builtin(__builtin_amdgcn_global_load_async_to_lds_b32)  && \
    __has_builtin(__builtin_amdgcn_s_wait_asynccnt)
#define HAVE_ASYNC 1
#else
#define HAVE_ASYNC 0
#endif

#if __has_builtin(__builtin_amdgcn_tensor_load_to_lds) && \
    __has_builtin(__builtin_amdgcn_s_wait_tensorcnt)
#define HAVE_TDM 1
#else
#define HAVE_TDM 0
#endif

#define B_TOT   4096
#define NODES   90
#define NPAD    96
#define INF_    195
#define FPAD    224      // 195 padded to 7*32
#define HID     128
#define OUTF    64
#define FLAT    5760     // 90*64
#define FC1N    512
#define FC2N    256
#define EPS     1e-5f

// ---------------------------------------------------------------- helpers

__device__ __forceinline__ v8f vzero8() {
    v8f z;
#pragma unroll
    for (int i = 0; i < 8; ++i) z[i] = 0.f;
    return z;
}

__device__ __forceinline__ v8f wmma16(v16h a, v16h b, v8f c) {
    return __builtin_amdgcn_wmma_f32_16x16x32_f16(
        false, a, false, b, (short)0, c, false, false);
}

// 16-bit A/B operand fragment: lanes 0-15 read k0+[0..7] & k0+[16..23],
// lanes 16-31 read k0+[8..15] & k0+[24..31] of row (row0 + lane%16).
__device__ __forceinline__ const _Float16* frag_ptr(const _Float16* base, int row0,
                                                    int ld, int k0, int lane) {
    int r = row0 + (lane & 15);
    int k = k0 + ((lane >> 4) << 3);
    return base + (size_t)r * ld + k;
}

__device__ __forceinline__ v16h load_frag(const _Float16* p) {
    v8h lo = *(const v8h*)(p);
    v8h hi = *(const v8h*)(p + 16);
    v16h r;
#pragma unroll
    for (int i = 0; i < 8; ++i) { r[i] = lo[i]; r[i + 8] = hi[i]; }
    return r;
}

// async 16-byte global->LDS copy (ASYNCcnt path), sync fallback otherwise
__device__ __forceinline__ void async_cp16(const _Float16* g, _Float16* l) {
#if HAVE_ASYNC
    __builtin_amdgcn_global_load_async_to_lds_b128(
        (g_v4i*)(uintptr_t)g, (l_v4i*)(uint32_t)(uintptr_t)l, 0, 0);
#else
    *(v8h*)l = *(const v8h*)g;
#endif
}

__device__ __forceinline__ void async_cp4(const float* g, float* l) {
#if HAVE_ASYNC
    __builtin_amdgcn_global_load_async_to_lds_b32(
        (g_i32*)(uintptr_t)g, (l_i32*)(uint32_t)(uintptr_t)l, 0, 0);
#else
    *l = *g;
#endif
}

template <int N>
__device__ __forceinline__ void wait_async() {
#if HAVE_ASYNC
    __builtin_amdgcn_s_wait_asynccnt(N);
#endif
}

// ---------------------------------------------------------------- kernel 0: f32 -> f16 with K padding

__global__ __launch_bounds__(256) void cvt_pad_f16(const float* __restrict__ src,
                                                   _Float16* __restrict__ dst,
                                                   int R, int K, int Kp) {
    int i = blockIdx.x * 256 + threadIdx.x;
    if (i >= R * Kp) return;
    int r = i / Kp, k = i - r * Kp;
    dst[i] = (k < K) ? (_Float16)src[(size_t)r * K + k] : (_Float16)0.f;
}

// ---------------------------------------------------------------- kernel 1: corr mask + 2x SAGE per sample

__global__ __launch_bounds__(256) void sage_kernel(
    const float*    __restrict__ x,     // [4096][90][195]
    const _Float16* __restrict__ W1l,   // [128][224] f16 padded
    const _Float16* __restrict__ W1r,   // [128][224]
    const float*    __restrict__ b1,    // [128]
    const _Float16* __restrict__ W2l,   // [64][128]
    const _Float16* __restrict__ W2r,   // [64][128]
    const float*    __restrict__ b2,    // [64]
    _Float16*       __restrict__ h2out) // [4096][5760]
{
    __shared__ __align__(16) float    sStage[NODES * INF_];  // raw f32 slab (TDM dest)
    __shared__ __align__(16) _Float16 sR0[NPAD * FPAD]; // agg1
    __shared__ __align__(16) _Float16 sR1[NPAD * FPAD]; // x row-major; later agg2 [96][128]
    __shared__ __align__(16) _Float16 sR2[FPAD * NPAD]; // xT [224][96]; later h1T [128][96]
    __shared__ __align__(16) _Float16 sM [NPAD * NPAD]; // mask (symmetric)
    __shared__ __align__(16) _Float16 sH1[NPAD * HID];  // h1 row-major
    __shared__ float sMu[NPAD], sRd[NPAD], sInvDeg[NPAD];

    const int tid  = threadIdx.x;
    const int lane = tid & 31;
    const int wv   = tid >> 5;
    const int b    = blockIdx.x;
    const float* xb = x + (size_t)b * NODES * INF_;

    // ---- Phase S: stage x[b] (90x195 f32) into LDS
#if HAVE_TDM
    if (wv == 0) {
        const uint32_t lds_off = (uint32_t)(uintptr_t)(void*)sStage;
        const unsigned long long ga = (unsigned long long)(uintptr_t)xb;
        // D# group0: count=1 | lds_addr | global_addr | type=2
        u32x4 g0 = { 1u,
                     lds_off,
                     (uint32_t)ga,
                     (uint32_t)((ga >> 32) & 0x1FFFFFFu) | (2u << 30) };
        // D# group1: data_size=4B(code 2); tensor_dim0=195; tensor_dim1=90;
        //            tile_dim0=195; tile_dim1=90; tensor_dim0_stride=195
        i32x8 g1 = { (int)(2u << 16),     // [17:16] data_size
                     (int)(195u << 16),   // [63:48] tensor_dim0[15:0]
                     (int)(90u  << 16),   // [79:64]=0, [95:80] tensor_dim1[15:0]
                     (int)(195u << 16),   // [111:96]=0, [127:112] tile_dim0
                     90,                  // [143:128] tile_dim1, [159:144] tile_dim2=0
                     195,                 // [191:160] tensor_dim0_stride lo
                     0, 0 };
        i32x4 gz = { 0, 0, 0, 0 };
#if defined(__clang_major__) && (__clang_major__ >= 23)
        i32x8 gz8 = { 0, 0, 0, 0, 0, 0, 0, 0 };
        __builtin_amdgcn_tensor_load_to_lds(g0, g1, gz, gz, gz8, 0);
#else
        __builtin_amdgcn_tensor_load_to_lds(g0, g1, gz, gz, 0);
#endif
        __builtin_amdgcn_s_wait_tensorcnt((short)0);
    }
#elif HAVE_ASYNC
    for (int i = tid; i < NODES * INF_; i += 256)
        async_cp4(xb + i, &sStage[i]);
    wait_async<0>();
#else
    for (int i = tid; i < NODES * INF_; i += 256)
        sStage[i] = xb[i];
#endif
    __syncthreads();

    // ---- Phase A: row mean + centered sumsq; write x (row-major f16) and xT
    if (tid < NPAD) {
        const int r = tid;
        float mu = 0.f, ss = 0.f;
        if (r < NODES) {
            const float* row = &sStage[r * INF_];
            float s = 0.f;
            for (int f = 0; f < INF_; ++f) s += row[f];
            mu = s * (1.0f / INF_);
            for (int f = 0; f < FPAD; ++f) {
                float v = (f < INF_) ? row[f] : 0.f;
                _Float16 hv = (_Float16)v;
                sR1[r * FPAD + f] = hv;
                sR2[f * NPAD + r] = hv;
                if (f < INF_) { float c = v - mu; ss += c * c; }
            }
        } else {
            for (int f = 0; f < FPAD; ++f) {
                sR1[r * FPAD + f] = (_Float16)0.f;
                sR2[f * NPAD + r] = (_Float16)0.f;
            }
        }
        sMu[r] = mu;
        sRd[r] = (ss > 0.f) ? rsqrtf(ss) : 0.f;
    }
    __syncthreads();

    // ---- Phase B: cov = x@xT (WMMA) - F*mu_i*mu_j; mask = corr > 0.5
    for (int t = wv; t < 36; t += 8) {
        const int ti = t / 6, tj = t % 6;
        v8f acc = vzero8();
        for (int kc = 0; kc < FPAD / 32; ++kc) {
            v16h a  = load_frag(frag_ptr(sR1, ti * 16, FPAD, kc * 32, lane));
            v16h bb = load_frag(frag_ptr(sR1, tj * 16, FPAD, kc * 32, lane));
            acc = wmma16(a, bb, acc);
        }
        const int col   = tj * 16 + (lane & 15);
        const int rbase = ti * 16 + ((lane >> 4) << 3);
        const float rdc = sRd[col], muc = sMu[col];
#pragma unroll
        for (int v = 0; v < 8; ++v) {
            const int row = rbase + v;
            float corr = (acc[v] - (float)INF_ * sMu[row] * muc) * sRd[row] * rdc;
            sM[row * NPAD + col] = (corr > 0.5f) ? (_Float16)1.f : (_Float16)0.f;
        }
    }
    __syncthreads();

    // ---- degrees (mask symmetric => maskT row sum == mask row sum)
    if (tid < NPAD) {
        float s = 0.f;
        for (int j = 0; j < NPAD; ++j) s += (float)sM[tid * NPAD + j];
        sInvDeg[tid] = (s > 0.f) ? (1.f / s) : 0.f;
    }
    __syncthreads();

    // ---- Phase C: agg1 = (mask @ x) * invdeg  -> sR0 [96][224]
    for (int t = wv; t < 6 * 14; t += 8) {
        const int ti = t / 14, tj = t % 14;
        v8f acc = vzero8();
        for (int kc = 0; kc < NPAD / 32; ++kc) {
            v16h a  = load_frag(frag_ptr(sM,  ti * 16, NPAD, kc * 32, lane));
            v16h bb = load_frag(frag_ptr(sR2, tj * 16, NPAD, kc * 32, lane));
            acc = wmma16(a, bb, acc);
        }
        const int col   = tj * 16 + (lane & 15);
        const int rbase = ti * 16 + ((lane >> 4) << 3);
#pragma unroll
        for (int v = 0; v < 8; ++v) {
            const int row = rbase + v;
            sR0[row * FPAD + col] = (_Float16)(acc[v] * sInvDeg[row]);
        }
    }
    __syncthreads();

    // ---- Phase D: h1 = relu(agg1@W1l^T + x@W1r^T + b1) -> sH1 and h1T -> sR2
    for (int t = wv; t < 6 * 8; t += 8) {
        const int ti = t / 8, tj = t % 8;
        v8f acc = vzero8();
        for (int kc = 0; kc < FPAD / 32; ++kc) {
            v16h a  = load_frag(frag_ptr(sR0, ti * 16, FPAD, kc * 32, lane));
            v16h bb = load_frag(frag_ptr(W1l, tj * 16, FPAD, kc * 32, lane));
            acc = wmma16(a, bb, acc);
            v16h a2  = load_frag(frag_ptr(sR1, ti * 16, FPAD, kc * 32, lane));
            v16h b2v = load_frag(frag_ptr(W1r, tj * 16, FPAD, kc * 32, lane));
            acc = wmma16(a2, b2v, acc);
        }
        const int col   = tj * 16 + (lane & 15);
        const float bi  = b1[col];
        const int rbase = ti * 16 + ((lane >> 4) << 3);
#pragma unroll
        for (int v = 0; v < 8; ++v) {
            const int row = rbase + v;
            float h = fmaxf(acc[v] + bi, 0.f);
            _Float16 hv = (_Float16)h;
            sH1[row * HID + col] = hv;
            sR2[col * NPAD + row] = hv;  // transposed copy for next B operand
        }
    }
    __syncthreads();

    // ---- Phase E: agg2 = (mask @ h1) * invdeg -> sR1 [96][128]
    for (int t = wv; t < 6 * 8; t += 8) {
        const int ti = t / 8, tj = t % 8;
        v8f acc = vzero8();
        for (int kc = 0; kc < NPAD / 32; ++kc) {
            v16h a  = load_frag(frag_ptr(sM,  ti * 16, NPAD, kc * 32, lane));
            v16h bb = load_frag(frag_ptr(sR2, tj * 16, NPAD, kc * 32, lane));
            acc = wmma16(a, bb, acc);
        }
        const int col   = tj * 16 + (lane & 15);
        const int rbase = ti * 16 + ((lane >> 4) << 3);
#pragma unroll
        for (int v = 0; v < 8; ++v) {
            const int row = rbase + v;
            sR1[row * HID + col] = (_Float16)(acc[v] * sInvDeg[row]);
        }
    }
    __syncthreads();

    // ---- Phase F: h2 = agg2@W2l^T + h1@W2r^T + b2 -> global (rows < 90 only)
    for (int t = wv; t < 6 * 4; t += 8) {
        const int ti = t / 4, tj = t % 4;
        v8f acc = vzero8();
        for (int kc = 0; kc < HID / 32; ++kc) {
            v16h a  = load_frag(frag_ptr(sR1, ti * 16, HID, kc * 32, lane));
            v16h bb = load_frag(frag_ptr(W2l, tj * 16, HID, kc * 32, lane));
            acc = wmma16(a, bb, acc);
            v16h a2  = load_frag(frag_ptr(sH1, ti * 16, HID, kc * 32, lane));
            v16h b2v = load_frag(frag_ptr(W2r, tj * 16, HID, kc * 32, lane));
            acc = wmma16(a2, b2v, acc);
        }
        const int col   = tj * 16 + (lane & 15);
        const float bi  = b2[col];
        const int rbase = ti * 16 + ((lane >> 4) << 3);
#pragma unroll
        for (int v = 0; v < 8; ++v) {
            const int row = rbase + v;
            if (row < NODES)
                h2out[(size_t)b * FLAT + row * OUTF + col] = (_Float16)(acc[v] + bi);
        }
    }
}

// ---------------------------------------------------------------- kernel 2: FC1 GEMM, LDS-staged + async double buffer

#define FC1_MT 64
#define FC1_NT 128
#define FC1_KC (FLAT / 32)   // 180 K-chunks

__global__ __launch_bounds__(256) void fc1_kernel(
    const _Float16* __restrict__ A,     // [4096][5760]
    const _Float16* __restrict__ W,     // [512][5760]
    const float* __restrict__ bias, const float* __restrict__ g,
    const float* __restrict__ be,   const float* __restrict__ mean,
    const float* __restrict__ var,
    _Float16* __restrict__ out)         // [4096][512]
{
    __shared__ __align__(16) _Float16 sA[2][FC1_MT * 32];  // 2 x 4KB
    __shared__ __align__(16) _Float16 sW[2][FC1_NT * 32];  // 2 x 8KB

    const int tid  = threadIdx.x;
    const int lane = tid & 31;
    const int wv   = tid >> 5;
    const int m0 = blockIdx.x * FC1_MT;
    const int n0 = blockIdx.y * FC1_NT;

    // copy assignments: A tile 64x32 (1 x b128/thread), W tile 128x32 (2 x b128/thread)
    const int arow = tid >> 2, acol = (tid & 3) * 8;
    const int wrow = tid >> 1, wcol = (tid & 1) * 16;
    const _Float16* gA = A + (size_t)(m0 + arow) * FLAT + acol;
    const _Float16* gW = W + (size_t)(n0 + wrow) * FLAT + wcol;

    // wave tile assignment: 4 m-tiles x 8 n-tiles = 32 tiles / 8 waves
    const int mi = wv & 3;
    const int nb = (wv >> 2) * 64;

    v8f acc[4] = { vzero8(), vzero8(), vzero8(), vzero8() };

    // prologue: stage chunk 0 into buffer 0
    {
        async_cp16(gA, &sA[0][arow * 32 + acol]);
        async_cp16(gW,     &sW[0][wrow * 32 + wcol]);
        async_cp16(gW + 8, &sW[0][wrow * 32 + wcol + 8]);
    }

    for (int kc = 0; kc < FC1_KC; ++kc) {
        const int cur = kc & 1;
        if (kc + 1 < FC1_KC) {
            const int nxt = cur ^ 1;
            const int ko = (kc + 1) * 32;
            async_cp16(gA + ko, &sA[nxt][arow * 32 + acol]);
            async_cp16(gW + ko,     &sW[nxt][wrow * 32 + wcol]);
            async_cp16(gW + ko + 8, &sW[nxt][wrow * 32 + wcol + 8]);
            wait_async<3>();   // retire oldest stage (in-order completion)
        } else {
            wait_async<0>();
        }
        __syncthreads();       // all waves' stage-kc data visible

        v16h a = load_frag(frag_ptr(&sA[cur][0], mi * 16, 32, 0, lane));
#pragma unroll
        for (int j = 0; j < 4; ++j) {
            v16h bb = load_frag(frag_ptr(&sW[cur][0], nb + j * 16, 32, 0, lane));
            acc[j] = wmma16(a, bb, acc[j]);
        }
        __syncthreads();       // done reading before buffer reuse
    }

    const int rb = m0 + mi * 16 + ((lane >> 4) << 3);
#pragma unroll
    for (int j = 0; j < 4; ++j) {
        const int col = n0 + nb + j * 16 + (lane & 15);
        const float bi = bias[col];
        const float sc = g[col] * rsqrtf(var[col] + EPS);
        const float mm = mean[col], bt = be[col];
#pragma unroll
        for (int v = 0; v < 8; ++v) {
            float h = fmaxf(acc[j][v] + bi, 0.f);
            h = (h - mm) * sc + bt;
            out[(size_t)(rb + v) * FC1N + col] = (_Float16)h;
        }
    }
}

// ---------------------------------------------------------------- kernel 3: FC2 GEMM + bias/relu/BN -> f32 "out"

__global__ __launch_bounds__(256) void fc2_kernel(
    const _Float16* __restrict__ A,     // [4096][512]
    const _Float16* __restrict__ W,     // [256][512]
    const float* __restrict__ bias, const float* __restrict__ g,
    const float* __restrict__ be,   const float* __restrict__ mean,
    const float* __restrict__ var,
    float* __restrict__ out)            // [4096][256]
{
    const int lane = threadIdx.x & 31, wv = threadIdx.x >> 5;
    const int m0 = blockIdx.x * 16;
    const int n0 = blockIdx.y * 128 + wv * 16;
    const _Float16* ap = A + (size_t)(m0 + (lane & 15)) * FC1N + ((lane >> 4) << 3);
    const _Float16* wp = W + (size_t)(n0 + (lane & 15)) * FC1N + ((lane >> 4) << 3);
    v8f acc = vzero8();
    for (int kc = 0; kc < FC1N / 32; ++kc) {
        __builtin_prefetch(wp + (kc + 4) * 32, 0, 0); // global_prefetch_b8
        v16h a  = load_frag(ap + kc * 32);
        v16h bb = load_frag(wp + kc * 32);
        acc = wmma16(a, bb, acc);
    }
    const int col = n0 + (lane & 15);
    const float bi = bias[col];
    const float sc = g[col] * rsqrtf(var[col] + EPS);
    const float mm = mean[col], bt = be[col];
    const int rb = m0 + ((lane >> 4) << 3);
#pragma unroll
    for (int v = 0; v < 8; ++v) {
        float h = fmaxf(acc[v] + bi, 0.f);
        out[(size_t)(rb + v) * FC2N + col] = (h - mm) * sc + bt;
    }
}

// ---------------------------------------------------------------- kernel 4: fc3 + softmax (wave-per-row, wave32)

__global__ __launch_bounds__(256) void head_kernel(
    const float* __restrict__ out256,   // [4096][256]
    const float* __restrict__ w,        // [2][256]
    const float* __restrict__ bvec,     // [2]
    float* __restrict__ soft)           // [4096][2]
{
    const int lane = threadIdx.x & 31, wv = threadIdx.x >> 5;
    const int row = blockIdx.x * 8 + wv;
    const float* o = out256 + (size_t)row * FC2N;
    float s0 = 0.f, s1 = 0.f;
    for (int i = lane; i < FC2N; i += 32) {
        float xv = o[i];
        s0 += xv * w[i];
        s1 += xv * w[FC2N + i];
    }
#pragma unroll
    for (int off = 16; off > 0; off >>= 1) {
        s0 += __shfl_xor(s0, off, 32);
        s1 += __shfl_xor(s1, off, 32);
    }
    if (lane == 0) {
        float l0 = s0 + bvec[0], l1 = s1 + bvec[1];
        float mx = fmaxf(l0, l1);
        float e0 = __expf(l0 - mx), e1 = __expf(l1 - mx);
        float inv = 1.f / (e0 + e1);
        soft[(size_t)row * 2 + 0] = e0 * inv;
        soft[(size_t)row * 2 + 1] = e1 * inv;
    }
}

// ---------------------------------------------------------------- host launch

extern "C" void kernel_launch(void* const* d_in, const int* in_sizes, int n_in,
                              void* d_out, int out_size, void* d_ws, size_t ws_size,
                              hipStream_t stream) {
    const float* x    = (const float*)d_in[0];
    const float* W1l  = (const float*)d_in[1];
    const float* b1   = (const float*)d_in[2];
    const float* W1r  = (const float*)d_in[3];
    const float* W2l  = (const float*)d_in[4];
    const float* b2   = (const float*)d_in[5];
    const float* W2r  = (const float*)d_in[6];
    const float* fc1w = (const float*)d_in[7];
    const float* fc1b = (const float*)d_in[8];
    const float* g1   = (const float*)d_in[9];
    const float* be1  = (const float*)d_in[10];
    const float* m1   = (const float*)d_in[11];
    const float* v1   = (const float*)d_in[12];
    const float* fc2w = (const float*)d_in[13];
    const float* fc2b = (const float*)d_in[14];
    const float* g2   = (const float*)d_in[15];
    const float* be2  = (const float*)d_in[16];
    const float* m2   = (const float*)d_in[17];
    const float* v2   = (const float*)d_in[18];
    const float* fc3w = (const float*)d_in[19];
    const float* fc3b = (const float*)d_in[20];

    // workspace layout (256B-aligned offsets)
    char* ws = (char*)d_ws;
    _Float16* w1l16  = (_Float16*)(ws + 0);                     // 128*224
    _Float16* w1r16  = (_Float16*)(ws + 57344);                 // 128*224
    _Float16* w2l16  = (_Float16*)(ws + 114688);                // 64*128
    _Float16* w2r16  = (_Float16*)(ws + 131072);                // 64*128
    _Float16* fc1w16 = (_Float16*)(ws + 147456);                // 512*5760
    _Float16* fc2w16 = (_Float16*)(ws + 6045696);               // 256*512
    _Float16* h2buf  = (_Float16*)(ws + 6307840);               // 4096*5760
    _Float16* act1   = (_Float16*)(ws + 53493760);              // 4096*512

    float* softm  = (float*)d_out;                              // [4096][2]
    float* out256 = (float*)d_out + (size_t)B_TOT * 2;          // [4096][256]

    auto cvt = [&](const float* s, _Float16* d, int R, int K, int Kp) {
        int n = R * Kp;
        cvt_pad_f16<<<dim3((n + 255) / 256), dim3(256), 0, stream>>>(s, d, R, K, Kp);
    };
    cvt(W1l,  w1l16,  HID,  INF_, FPAD);
    cvt(W1r,  w1r16,  HID,  INF_, FPAD);
    cvt(W2l,  w2l16,  OUTF, HID,  HID);
    cvt(W2r,  w2r16,  OUTF, HID,  HID);
    cvt(fc1w, fc1w16, FC1N, FLAT, FLAT);
    cvt(fc2w, fc2w16, FC2N, FC1N, FC1N);

    sage_kernel<<<dim3(B_TOT), dim3(256), 0, stream>>>(
        x, w1l16, w1r16, b1, w2l16, w2r16, b2, h2buf);

    fc1_kernel<<<dim3(B_TOT / FC1_MT, FC1N / FC1_NT), dim3(256), 0, stream>>>(
        h2buf, fc1w16, fc1b, g1, be1, m1, v1, act1);

    fc2_kernel<<<dim3(B_TOT / 16, FC2N / 128), dim3(256), 0, stream>>>(
        act1, fc2w16, fc2b, g2, be2, m2, v2, out256);

    head_kernel<<<dim3(B_TOT / 8), dim3(256), 0, stream>>>(
        out256, fc3w, fc3b, softm);
}
